// GCN_66425964200295
// MI455X (gfx1250) — matrix operand order repaired and checked by reading, not compiled, not run on hardware
//
#include <hip/hip_runtime.h>
#include <hip/hip_bf16.h>

// ---------------------------------------------------------------------------
// GCN (2x GCNConv mean + classifier) for MI455X (gfx1250, wave32, WMMA).
// GEMMs: v_wmma_f32_16x16x32_bf16 with operands pre-swizzled into hardware
// fragment order (per-lane 32B contiguous -> global_load_b128 pairs).
// Propagation: edge-parallel wave32 + global_atomic_add_f32 (hW fits in L2).
// ---------------------------------------------------------------------------

typedef __attribute__((ext_vector_type(16))) __bf16 v16bf;
typedef __attribute__((ext_vector_type(8)))  float  v8f;

#define FEAT 128   // H1 == H2 == 128
#define INF  256   // input feature dim
#define COUT 10    // classifier outputs

__device__ __forceinline__ unsigned short f2bf(float f) {
    unsigned int u = __float_as_uint(f);
    unsigned int r = 0x7FFFu + ((u >> 16) & 1u);   // round-to-nearest-even
    return (unsigned short)((u + r) >> 16);
}

// ---- fragment-order ("swizzled") element index for 16-bit WMMA operands ----
// A operand (16x32 tile of row-major [M,K]):
//   lane = (k.bit3)<<4 | (m&15);  vgpr v = (k.bit4)<<2 | k.bits2:1;  pair = k.bit0
__device__ __forceinline__ size_t idxA(int m, int k, int K) {
    int mtile = m >> 4, l16 = m & 15;
    int kbi = k >> 5, kk = k & 31;
    int v    = (((kk >> 4) & 1) << 2) | ((kk >> 1) & 3);
    int lane = (((kk >> 3) & 1) << 4) | l16;
    return (((size_t)mtile * (K >> 5) + kbi) * 32 + lane) * 16 + (v << 1) + (kk & 1);
}
// B operand (32x16 tile of row-major [K,NB]):
//   lane = (k.bit3)<<4 | (n&15);  vgpr v = (k.bit4)<<2 | k.bits2:1;  pair = k.bit0
__device__ __forceinline__ size_t idxB(int k, int n, int K) {
    int ntile = n >> 4, l16 = n & 15;
    int kbi = k >> 5, kk = k & 31;
    int v    = (((kk >> 4) & 1) << 2) | ((kk >> 1) & 3);
    int lane = (((kk >> 3) & 1) << 4) | l16;
    return (((size_t)ntile * (K >> 5) + kbi) * 32 + lane) * 16 + (v << 1) + (kk & 1);
}

// ---------------- utility kernels ----------------

__global__ void zero_f32_kernel(float* __restrict__ p, long long n) {
    long long i = (long long)blockIdx.x * blockDim.x + threadIdx.x;
    if (i < n) p[i] = 0.0f;
}

// f32 row-major [M,K] -> bf16 A-fragment order
__global__ void cvt_swzA_kernel(const float* __restrict__ in,
                                unsigned short* __restrict__ out,
                                long long n, int K) {
    long long i = (long long)blockIdx.x * blockDim.x + threadIdx.x;
    if (i < n) {
        int m = (int)(i / K), k = (int)(i % K);
        out[idxA(m, k, K)] = f2bf(in[i]);
    }
}

// f32 row-major [K,NSRC] -> bf16 B-fragment order, zero-padded to NB columns
__global__ void cvt_swzB_kernel(const float* __restrict__ in,
                                unsigned short* __restrict__ out,
                                int K, int NB, int NSRC) {
    int i = blockIdx.x * blockDim.x + threadIdx.x;
    if (i < K * NB) {
        int k = i / NB, nc = i % NB;
        float v = (nc < NSRC) ? in[k * NSRC + nc] : 0.0f;
        out[idxB(k, nc, K)] = f2bf(v);
    }
}

// deg[dst] += 1 for every real edge (self loops folded in later)
__global__ void degree_kernel(const long long* __restrict__ dst,
                              float* __restrict__ deg, int E) {
    int e = blockIdx.x * blockDim.x + threadIdx.x;
    if (e < E) atomicAdd(&deg[(int)dst[e]], 1.0f);
}

// dinv = rsqrt(deg+1); invdeg = 1/(deg+1)   (deg+1 >= 1 always)
__global__ void dinv_kernel(const float* __restrict__ deg,
                            float* __restrict__ dinv,
                            float* __restrict__ invdeg, int n) {
    int i = blockIdx.x * blockDim.x + threadIdx.x;
    if (i < n) {
        float d = deg[i] + 1.0f;
        dinv[i]   = rsqrtf(d);
        invdeg[i] = 1.0f / d;
    }
}

// ---------------- WMMA GEMM over pre-swizzled operands ----------------
// One wave computes a 16x16 tile. blockDim.x = 32*(NB/16); blockIdx.x = M tile.
// Inner loop: 2x b128 A-load + 2x b128 B-load + 1 wmma, fully unrolled.
template <int K, int NB, bool CLS>
__global__ void gemm_wmma_bf16_kernel(const unsigned short* __restrict__ A,
                                      const unsigned short* __restrict__ B,
                                      float* __restrict__ C,
                                      const float* __restrict__ bias) {
    constexpr int KB = K / 32;
    const int wave = threadIdx.x >> 5;
    const int lane = threadIdx.x & 31;
    const int half = lane >> 4;
    const int l16  = lane & 15;
    const int m0   = blockIdx.x << 4;
    const int ncol = (wave << 4) + l16;

    const char* Abase = (const char*)A + (((size_t)blockIdx.x * KB) * 32 + lane) * 32;
    const char* Bbase = (const char*)B + (((size_t)wave * KB) * 32 + lane) * 32;

    v8f acc = {};
#pragma unroll
    for (int kbi = 0; kbi < KB; ++kbi) {
        union { v16bf v; uint4 q[2]; } a, b;
        a.q[0] = *(const uint4*)(Abase + kbi * 1024);
        a.q[1] = *(const uint4*)(Abase + kbi * 1024 + 16);
        b.q[0] = *(const uint4*)(Bbase + kbi * 1024);
        b.q[1] = *(const uint4*)(Bbase + kbi * 1024 + 16);
        acc = __builtin_amdgcn_wmma_f32_16x16x32_bf16(
            false, a.v, false, b.v, (short)0, acc, false, false);
    }

    if (!CLS) {
        // C/D layout: VGPR v -> row m0 + v + 8*half, col = ncol  (no masks:
        // M is a multiple of 16 and NOUT == NB)
#pragma unroll
        for (int v = 0; v < 8; ++v)
            C[(size_t)(m0 + v + (half << 3)) * NB + ncol] = acc[v];
    } else {
        const bool wr = ncol < COUT;
        const float bv = wr ? bias[ncol] : 0.0f;   // hoisted single load
#pragma unroll
        for (int v = 0; v < 8; ++v)
            if (wr)
                C[(size_t)(m0 + v + (half << 3)) * COUT + ncol] = acc[v] + bv;
    }
}

// ---------------- edge-parallel propagation (one wave per edge) ----------------
// agg[dst] += t[src] * dinv[src]*dinv[dst]   across 128 features
__global__ void propagate_kernel(const long long* __restrict__ src,
                                 const long long* __restrict__ dst,
                                 const float* __restrict__ t,
                                 const float* __restrict__ dinv,
                                 float* __restrict__ agg, int E) {
    int e = blockIdx.x * (blockDim.x >> 5) + (threadIdx.x >> 5);
    if (e >= E) return;
    const int lane = threadIdx.x & 31;
    const int s = (int)src[e];
    const int d = (int)dst[e];
    const float norm = dinv[s] * dinv[d];
    const float4 v = ((const float4*)(t + (size_t)s * FEAT))[lane];
    float* arow = agg + (size_t)d * FEAT + lane * 4;
    atomicAdd(arow + 0, v.x * norm);
    atomicAdd(arow + 1, v.y * norm);
    atomicAdd(arow + 2, v.z * norm);
    atomicAdd(arow + 3, v.w * norm);
}

// ---- fused: self-loop + mean + bias + ReLU, writing bf16 in A-fragment order ----
__global__ void finalize_kernel(const float* __restrict__ t,
                                const float* __restrict__ agg,
                                const float* __restrict__ dinv,
                                const float* __restrict__ invdeg,
                                const float* __restrict__ bias,
                                unsigned short* __restrict__ hout, int n) {
    const int i = blockIdx.x;      // node
    const int f = threadIdx.x;     // 0..127
    if (i >= n) return;
    const float di = dinv[i];
    const size_t idx = (size_t)i * FEAT + f;
    float v = (agg[idx] + t[idx] * di * di) * invdeg[i] + bias[f];
    hout[idxA(i, f, FEAT)] = f2bf(fmaxf(v, 0.0f));
}

// ---------------------------------------------------------------------------

extern "C" void kernel_launch(void* const* d_in, const int* in_sizes, int n_in,
                              void* d_out, int out_size, void* d_ws, size_t ws_size,
                              hipStream_t stream) {
    const float*     x  = (const float*)d_in[0];
    const long long* ei = (const long long*)d_in[1];
    const float*     W1 = (const float*)d_in[2];
    const float*     b1 = (const float*)d_in[3];
    const float*     W2 = (const float*)d_in[4];
    const float*     b2 = (const float*)d_in[5];
    const float*     Wc = (const float*)d_in[6];
    const float*     bc = (const float*)d_in[7];
    float* out = (float*)d_out;

    const int N = in_sizes[0] / INF;    // 50000 (multiple of 16)
    const int E = in_sizes[1] / 2;      // 800000
    const long long* srcp = ei;
    const long long* dstp = ei + E;

    // ---- carve workspace (256B aligned chunks) ----
    char* wsp = (char*)d_ws;
    auto carve = [&](size_t bytes) -> char* {
        char* p = wsp;
        wsp += (bytes + 255) & ~(size_t)255;
        return p;
    };
    float*          deg    = (float*)         carve((size_t)N * 4);
    float*          dinv   = (float*)         carve((size_t)N * 4);
    float*          invdeg = (float*)         carve((size_t)N * 4);
    unsigned short* xs     = (unsigned short*)carve((size_t)N * INF * 2);   // A-swz x
    unsigned short* w1s    = (unsigned short*)carve((size_t)INF * FEAT * 2);// B-swz W1
    unsigned short* w2s    = (unsigned short*)carve((size_t)FEAT * FEAT * 2);
    unsigned short* wcs    = (unsigned short*)carve((size_t)FEAT * 16 * 2); // padded
    float*          t      = (float*)         carve((size_t)N * FEAT * 4);
    float*          agg    = (float*)         carve((size_t)N * FEAT * 4);
    unsigned short* hs     = (unsigned short*)carve((size_t)N * FEAT * 2);  // A-swz h

    const long long nNF = (long long)N * FEAT;

    // ---- degrees / norms ----
    zero_f32_kernel<<<(N + 255) / 256, 256, 0, stream>>>(deg, N);
    degree_kernel<<<(E + 255) / 256, 256, 0, stream>>>(dstp, deg, E);
    dinv_kernel<<<(N + 255) / 256, 256, 0, stream>>>(deg, dinv, invdeg, N);

    // ---- bf16 conversions into fragment order ----
    {
        long long nx = (long long)N * INF;
        cvt_swzA_kernel<<<(unsigned)((nx + 255) / 256), 256, 0, stream>>>(x, xs, nx, INF);
    }
    cvt_swzB_kernel<<<(INF * FEAT + 255) / 256, 256, 0, stream>>>(W1, w1s, INF, FEAT, FEAT);
    cvt_swzB_kernel<<<(FEAT * FEAT + 255) / 256, 256, 0, stream>>>(W2, w2s, FEAT, FEAT, FEAT);
    cvt_swzB_kernel<<<(FEAT * 16 + 255) / 256, 256, 0, stream>>>(Wc, wcs, FEAT, 16, COUT);

    const int mtiles  = (N + 15) / 16;   // 3125 (exact)
    const int eblocks = (E + 7) / 8;     // 8 edges / 256-thread block

    // ---- layer 1 ----
    gemm_wmma_bf16_kernel<INF, FEAT, false><<<mtiles, 32 * (FEAT / 16), 0, stream>>>(
        xs, w1s, t, nullptr);
    zero_f32_kernel<<<(unsigned)((nNF + 255) / 256), 256, 0, stream>>>(agg, nNF);
    propagate_kernel<<<eblocks, 256, 0, stream>>>(srcp, dstp, t, dinv, agg, E);
    finalize_kernel<<<N, FEAT, 0, stream>>>(t, agg, dinv, invdeg, b1, hs, N);

    // ---- layer 2 ----
    gemm_wmma_bf16_kernel<FEAT, FEAT, false><<<mtiles, 32 * (FEAT / 16), 0, stream>>>(
        hs, w2s, t, nullptr);
    zero_f32_kernel<<<(unsigned)((nNF + 255) / 256), 256, 0, stream>>>(agg, nNF);
    propagate_kernel<<<eblocks, 256, 0, stream>>>(srcp, dstp, t, dinv, agg, E);
    finalize_kernel<<<N, FEAT, 0, stream>>>(t, agg, dinv, invdeg, b2, hs, N);

    // ---- classifier: [N,128] x [128,16(pad)] -> [N,10] with bias ----
    gemm_wmma_bf16_kernel<FEAT, 16, true><<<mtiles, 32, 0, stream>>>(
        hs, wcs, out, bc);
}
